// ConditionalEdgeDenoiser_73358041416219
// MI455X (gfx1250) — matrix operand order repaired and checked by reading, not compiled, not run on hardware
//
#include <hip/hip_runtime.h>
#include <hip/hip_bf16.h>

// CDNA5 / gfx1250 fused ConditionalEdgeDenoiser.
// 3-layer MLP over 262144 rows, bf16 WMMA (v_wmma_f32_16x16x32_bf16) with f32 accum.

typedef __attribute__((ext_vector_type(16))) __bf16 v16bf;
typedef __attribute__((ext_vector_type(8)))  float  v8f;

// ---- ISA fragment layouts (CDNA5 ISA §7.12.2) ----
// A (16x32 bf16): lane L holds row m = L&15; VGPR v holds K pair:
//   v<4 : K = 8*(L>>4) + 2v        (K 0..7 lanes 0-15, 8..15 lanes 16-31)
//   v>=4: K = 16 + 8*(L>>4) + 2(v-4)
__device__ __forceinline__ int frag_k_a(int v, int halfk) {
  return (v < 4) ? (halfk * 8 + v * 2) : (16 + halfk * 8 + (v - 4) * 2);
}
// B (32x16 bf16): lane L holds col n = L&15; VGPR v holds K pair (same as A):
//   v<4 : K = 8*(L>>4) + 2v;  v>=4: K = 16 + 8*(L>>4) + 2(v-4)
__device__ __forceinline__ int frag_k_b(int v, int halfk) {
  return (v < 4) ? (halfk * 8 + v * 2) : (16 + halfk * 8 + (v - 4) * 2);
}

// Pack a K x Ncols row-major f32 weight matrix into bf16 WMMA B-fragments.
// Block layout: blk = nt*KT + kt, 512 bf16 per block; lane L's 16 values at
// blk*512 + L*16 so the main kernel loads each fragment as one 32B load.
__global__ void pack_w_bf16(const float* __restrict__ W, int Krows, int Ncols,
                            int KT, int NT, __bf16* __restrict__ dst) {
  int gid  = blockIdx.x * blockDim.x + threadIdx.x;
  int blk  = gid >> 5;
  int lane = gid & 31;
  if (blk >= KT * NT) return;
  int nt    = blk / KT;
  int kt    = blk % KT;
  int halfk = lane >> 4;
  int n     = nt * 16 + (lane & 15);
  __bf16* out = dst + (size_t)blk * 512 + lane * 16;
#pragma unroll
  for (int v = 0; v < 8; ++v) {
    int k = kt * 32 + frag_k_b(v, halfk);
#pragma unroll
    for (int e = 0; e < 2; ++e) {
      int kk = k + e;
      float val = (kk < Krows && n < Ncols) ? W[(size_t)kk * Ncols + n] : 0.0f;
      out[v * 2 + e] = (__bf16)val;
    }
  }
}

// Fused MLP: each workgroup = 128 rows (fixed b,i; j = j0..j0+127), 8 waves x 16 rows.
__global__ __launch_bounds__(256) void edge_denoiser_kernel(
    const float* __restrict__ edge_t_, const float* __restrict__ x_cond,
    const float* __restrict__ code_cond, const int* __restrict__ t,
    const unsigned char* __restrict__ node_mask,
    const float* __restrict__ b1, const float* __restrict__ b2,
    const float* __restrict__ b3,
    const __bf16* __restrict__ W1p, const __bf16* __restrict__ W2p,
    const __bf16* __restrict__ W3p, float* __restrict__ out) {
  extern __shared__ char smem_raw[];
  __bf16* Xs = (__bf16*)smem_raw;                       // [128][320] bf16, later reused as H2 [128][256]
  __bf16* Hs = (__bf16*)(smem_raw + 128 * 320 * 2);     // [128][256] bf16

  const int tid     = threadIdx.x;
  const int wave    = tid >> 5;
  const int lane    = tid & 31;
  const int baseRow = blockIdx.x * 128;
  const int b       = baseRow >> 16;       // N*N = 65536 rows per batch
  const int within  = baseRow & 65535;
  const int i       = within >> 8;
  const int j0      = within & 255;        // 0 or 128

  // ---- Build X tile [128 x 320] in LDS (292 real cols + 28 zero pad) ----
  const float tval = (float)t[b];
  const float lnf  = __logf(10000.0f) / 63.0f;   // half-1 = 63
  for (int idx = tid; idx < 128 * 320; idx += 256) {
    int m = idx / 320;
    int c = idx - m * 320;
    float v = 0.0f;
    if (c < 4) {                                         // edge_t[b,i,j,c]
      v = edge_t_[((((size_t)b * 256 + i) * 256) + (j0 + m)) * 4 + c];
    } else if (c < 164) {                                // src ctx (i) / dst ctx (j)
      int col  = (c < 84) ? (c - 4) : (c - 84);
      int node = (c < 84) ? i : (j0 + m);
      v = (col < 64) ? x_cond[((size_t)b * 256 + node) * 64 + col]
                     : code_cond[((size_t)b * 256 + node) * 16 + (col - 64)];
    } else if (c < 292) {                                // sinusoidal time embedding
      int c2 = c - 164;
      int k  = (c2 < 64) ? c2 : (c2 - 64);
      float a = tval * __expf(-(float)k * lnf);
      v = (c2 < 64) ? __sinf(a) : __cosf(a);
    }
    Xs[idx] = (__bf16)v;
  }
  __syncthreads();

  const int halfk = lane >> 4;
  const int nlow  = lane & 15;
  const int mbase = wave * 16;
  const int mrow  = mbase + nlow;   // A-fragment row for this lane

  // ---- Layer 1: [16x320] @ [320x256] + b1, SiLU -> Hs ----
  for (int nt = 0; nt < 16; ++nt) {
    v8f acc = {};
#pragma unroll
    for (int kt = 0; kt < 10; ++kt) {
      union { v16bf v; unsigned int u[8]; } A;
#pragma unroll
      for (int vv = 0; vv < 8; ++vv) {
        int k = kt * 32 + frag_k_a(vv, halfk);
        A.u[vv] = *(const unsigned int*)(Xs + mrow * 320 + k);
      }
      v16bf Bf = *(const v16bf*)(W1p + (size_t)(nt * 10 + kt) * 512 + lane * 16);
      acc = __builtin_amdgcn_wmma_f32_16x16x32_bf16(false, A.v, false, Bf,
                                                    (short)0, acc, false, false);
    }
    int   n    = nt * 16 + nlow;
    float bias = b1[n];
#pragma unroll
    for (int r = 0; r < 8; ++r) {
      int   m = mbase + r + 8 * halfk;
      float x = acc[r] + bias;
      float s = x / (1.0f + __expf(-x));                 // SiLU
      Hs[m * 256 + n] = (__bf16)s;
    }
  }
  __syncthreads();

  // ---- Layer 2: [16x256] @ [256x256] + b2, SiLU -> H2 (reuse Xs) ----
  for (int nt = 0; nt < 16; ++nt) {
    v8f acc = {};
#pragma unroll
    for (int kt = 0; kt < 8; ++kt) {
      union { v16bf v; unsigned int u[8]; } A;
#pragma unroll
      for (int vv = 0; vv < 8; ++vv) {
        int k = kt * 32 + frag_k_a(vv, halfk);
        A.u[vv] = *(const unsigned int*)(Hs + mrow * 256 + k);
      }
      v16bf Bf = *(const v16bf*)(W2p + (size_t)(nt * 8 + kt) * 512 + lane * 16);
      acc = __builtin_amdgcn_wmma_f32_16x16x32_bf16(false, A.v, false, Bf,
                                                    (short)0, acc, false, false);
    }
    int   n    = nt * 16 + nlow;
    float bias = b2[n];
#pragma unroll
    for (int r = 0; r < 8; ++r) {
      int   m = mbase + r + 8 * halfk;
      float x = acc[r] + bias;
      float s = x / (1.0f + __expf(-x));
      Xs[m * 256 + n] = (__bf16)s;
    }
  }
  __syncthreads();

  // ---- Layer 3: [16x256] @ [256x16(4 valid)] + b3, mask -> out ----
  {
    v8f acc = {};
#pragma unroll
    for (int kt = 0; kt < 8; ++kt) {
      union { v16bf v; unsigned int u[8]; } A;
#pragma unroll
      for (int vv = 0; vv < 8; ++vv) {
        int k = kt * 32 + frag_k_a(vv, halfk);
        A.u[vv] = *(const unsigned int*)(Xs + mrow * 256 + k);
      }
      v16bf Bf = *(const v16bf*)(W3p + (size_t)kt * 512 + lane * 16);
      acc = __builtin_amdgcn_wmma_f32_16x16x32_bf16(false, A.v, false, Bf,
                                                    (short)0, acc, false, false);
    }
    if (nlow < 4) {
      float         bias = b3[nlow];
      unsigned char mi   = node_mask[b * 256 + i];
#pragma unroll
      for (int r = 0; r < 8; ++r) {
        int m = mbase + r + 8 * halfk;
        int j = j0 + m;
        unsigned char mj = node_mask[b * 256 + j];
        float val = (mi && mj) ? (acc[r] + bias) : 0.0f;
        out[((((size_t)b * 256 + i) * 256) + j) * 4 + nlow] = val;
      }
    }
  }
}

extern "C" void kernel_launch(void* const* d_in, const int* in_sizes, int n_in,
                              void* d_out, int out_size, void* d_ws, size_t ws_size,
                              hipStream_t stream) {
  (void)in_sizes; (void)n_in; (void)out_size; (void)ws_size;
  const float*         edge_t_   = (const float*)d_in[0];
  const float*         x_cond    = (const float*)d_in[1];
  const float*         code_cond = (const float*)d_in[2];
  const int*           t         = (const int*)d_in[3];
  const unsigned char* node_mask = (const unsigned char*)d_in[4];
  const float*         W1        = (const float*)d_in[5];
  const float*         b1        = (const float*)d_in[6];
  const float*         W2        = (const float*)d_in[7];
  const float*         b2        = (const float*)d_in[8];
  const float*         W3        = (const float*)d_in[9];
  const float*         b3        = (const float*)d_in[10];
  float*               out       = (float*)d_out;

  // Workspace: packed bf16 weight fragments (~303 KB), L2-resident.
  __bf16* W1p = (__bf16*)d_ws;                                   // 10 KT x 16 NT = 160 blk (163840 B)
  __bf16* W2p = (__bf16*)((char*)d_ws + 163840);                 //  8 KT x 16 NT = 128 blk (131072 B)
  __bf16* W3p = (__bf16*)((char*)d_ws + 163840 + 131072);        //  8 KT x  1 NT =   8 blk (  8192 B)

  pack_w_bf16<<<(160 * 32 + 255) / 256, 256, 0, stream>>>(W1, 292, 256, 10, 16, W1p);
  pack_w_bf16<<<(128 * 32 + 255) / 256, 256, 0, stream>>>(W2, 256, 256,  8, 16, W2p);
  pack_w_bf16<<<(  8 * 32 + 255) / 256, 256, 0, stream>>>(W3, 256,   4,  8,  1, W3p);

  const size_t smem = 128 * 320 * 2 + 128 * 256 * 2;   // 144 KB LDS per workgroup
  edge_denoiser_kernel<<<2048, 256, smem, stream>>>(
      edge_t_, x_cond, code_cond, t, node_mask, b1, b2, b3, W1p, W2p, W3p, out);
}
